// MultiHeadGeometryAttention_4672924418747
// MI455X (gfx1250) — matrix operand order repaired and checked by reading, not compiled, not run on hardware
//
#include <hip/hip_runtime.h>
#include <hip/hip_bf16.h>
#include <stdint.h>

#define Bc 4
#define Hc 16
#define Dc 64
#define NQc 1024
#define NKc 1024
#define DM 1024     // D_MODEL == H*DK == H*DV
#define LN_EPS 1e-5f

typedef __attribute__((ext_vector_type(16))) __bf16     v16bf;
typedef __attribute__((ext_vector_type(8)))  float      v8f;
typedef __attribute__((ext_vector_type(8)))  unsigned   v8u;

// ---------------- helpers ----------------

__device__ __forceinline__ unsigned short f2bf(float f) {
    unsigned u = __float_as_uint(f);
    return (unsigned short)((u + 0x7FFFu + ((u >> 16) & 1u)) >> 16);
}

// Contiguous 32B load of 16 bf16 -> WMMA B-fragment (per-lane K-contiguous layout)
__device__ __forceinline__ v16bf load16(const unsigned short* p) {
    v8u t = *(const v8u*)p;
    return __builtin_bit_cast(v16bf, t);
}

// A-fragment (16x32 bf16): per lane, two contiguous 16B loads:
//   elems 0..7  -> K = 16*hl + e      (at p + 16*hl)
//   elems 8..15 -> K = 16*hl + 8 + e  (at p + 16*hl + 8)
__device__ __forceinline__ v16bf load_afrag(const unsigned short* p, int hl) {
    uint4 lo = *(const uint4*)(p + 16 * hl);
    uint4 hi = *(const uint4*)(p + 16 * hl + 8);
    v8u t;
    t[0] = lo.x; t[1] = lo.y; t[2] = lo.z; t[3] = lo.w;
    t[4] = hi.x; t[5] = hi.y; t[6] = hi.z; t[7] = hi.w;
    return __builtin_bit_cast(v16bf, t);
}

__device__ __forceinline__ v8f wmma_bf16(v16bf a, v16bf b, v8f c) {
    return __builtin_amdgcn_wmma_f32_16x16x32_bf16(
        false, a, false, b, (short)0, c, false, false);
}

// ---------------- elementwise fp32 -> bf16 ----------------

__global__ __launch_bounds__(256) void cvt_bf16_kernel(const float* __restrict__ in,
                                                       unsigned short* __restrict__ out,
                                                       int n) {
    int i = (blockIdx.x * 256 + threadIdx.x) * 4;
    if (i < n) {
        float4 f = *(const float4*)(in + i);
        out[i + 0] = f2bf(f.x);
        out[i + 1] = f2bf(f.y);
        out[i + 2] = f2bf(f.z);
        out[i + 3] = f2bf(f.w);
    }
}

// ---------------- fp32 [K,N] -> bf16 [N,K] transpose-convert (weights) ----------------

__global__ __launch_bounds__(256) void transposeW_kernel(const float* __restrict__ W,
                                                         unsigned short* __restrict__ Wt,
                                                         int Kd, int Nd) {
    __shared__ float tile[32][33];
    int k0 = blockIdx.y * 32, n0 = blockIdx.x * 32;
    int tx = threadIdx.x, ty = threadIdx.y;   // 32 x 8
    #pragma unroll
    for (int i = 0; i < 32; i += 8)
        tile[ty + i][tx] = W[(size_t)(k0 + ty + i) * Nd + n0 + tx];
    __syncthreads();
    #pragma unroll
    for (int i = 0; i < 32; i += 8)
        Wt[(size_t)(n0 + ty + i) * Kd + k0 + tx] = f2bf(tile[tx][ty + i]);
}

// ---------------- bf16 [b][key][feat] -> bf16 [b][feat][key] (V transpose) ----------------

__global__ __launch_bounds__(256) void transposeV_kernel(const unsigned short* __restrict__ V,
                                                         unsigned short* __restrict__ Vt) {
    __shared__ unsigned short tile[32][34];
    int b = blockIdx.z;
    int r0 = blockIdx.y * 32;   // key
    int c0 = blockIdx.x * 32;   // feature
    int tx = threadIdx.x, ty = threadIdx.y;
    #pragma unroll
    for (int i = 0; i < 32; i += 8)
        tile[ty + i][tx] = V[((size_t)b * NKc + r0 + ty + i) * DM + c0 + tx];
    __syncthreads();
    #pragma unroll
    for (int i = 0; i < 32; i += 8)
        Vt[((size_t)b * DM + c0 + ty + i) * NKc + r0 + tx] = tile[tx][ty + i];
}

// ---------------- WMMA GEMM: C[M,N] = A[M,K](bf16) * Bt[N,K](bf16) + bias ----------------
// 128 threads = 4 waves; block tile 64x64; each wave: 16 rows x 64 cols (4 accs).

template <int OUT_BF16>
__global__ __launch_bounds__(128) void gemm_bf16_kernel(const unsigned short* __restrict__ A,
                                                        const unsigned short* __restrict__ Bt,
                                                        const float* __restrict__ bias,
                                                        void* __restrict__ C,
                                                        int M, int N, int K) {
    const int lane = threadIdx.x & 31;
    const int wave = threadIdx.x >> 5;
    const int hl = lane >> 4;
    const int ln = lane & 15;
    const int mbase = blockIdx.y * 64 + wave * 16;
    const int nbase = blockIdx.x * 64;

    v8f acc[4] = {};
    const unsigned short* arow  = A + (size_t)(mbase + ln) * K;
    const unsigned short* bcol0 = Bt + (size_t)(nbase + ln) * K + 16 * hl;

    for (int kc = 0; kc < K; kc += 32) {
        __builtin_prefetch(arow + kc + 128, 0, 0);            // global_prefetch_b8
        __builtin_prefetch(bcol0 + kc + 128, 0, 0);
        v16bf a = load_afrag(arow + kc, hl);
        #pragma unroll
        for (int t = 0; t < 4; ++t) {
            v16bf bf = load16(bcol0 + (size_t)t * 16 * K + kc);
            acc[t] = wmma_bf16(a, bf, acc[t]);
        }
    }

    #pragma unroll
    for (int t = 0; t < 4; ++t) {
        int n = nbase + t * 16 + ln;
        float bv = bias[n];
        #pragma unroll
        for (int r = 0; r < 8; ++r) {
            int m = mbase + r + 8 * hl;
            float v = acc[t][r] + bv;
            if (OUT_BF16)
                ((unsigned short*)C)[(size_t)m * N + n] = f2bf(v);
            else
                ((float*)C)[(size_t)m * N + n] = v;
        }
    }
}

// ---------------- fused flash attention with geometry log-bias + mask ----------------
// grid (NQ/64, H, B), 128 threads = 4 waves; wave handles 16 q-rows, streams 32 keys/iter.

__global__ __launch_bounds__(128) void attn_kernel(const unsigned short* __restrict__ qb,  // [B,NQ,DM]
                                                   const unsigned short* __restrict__ kb,  // [B,NK,DM]
                                                   const unsigned short* __restrict__ vt,  // [B,DM,NK]
                                                   const float* __restrict__ geom,         // [B,H,NQ,NK]
                                                   const unsigned char* __restrict__ mask, // [B,H,NQ,NK]
                                                   unsigned short* __restrict__ ob) {      // [B,NQ,DM]
    __shared__ unsigned short plds[4][16 * 32];   // per-wave P staging (row-major 16x32)
    const int lane = threadIdx.x & 31;
    const int wave = threadIdx.x >> 5;
    const int hl = lane >> 4;
    const int ln = lane & 15;
    const int b = blockIdx.z, h = blockIdx.y;
    const int qbase = blockIdx.x * 64 + wave * 16;
    const float NEG_INF = -__builtin_inff();

    // resident Q fragments (two 32-wide d chunks of DK=64)
    v16bf aq[2];
    {
        const unsigned short* qrow = qb + ((size_t)(b * NQc + qbase + ln)) * DM + h * Dc;
        aq[0] = load_afrag(qrow, hl);
        aq[1] = load_afrag(qrow + 32, hl);
    }

    v8f acc0 = {}, acc1 = {}, acc2 = {}, acc3 = {};
    float mrow[8], lrow[8];
    #pragma unroll
    for (int r = 0; r < 8; ++r) { mrow[r] = NEG_INF; lrow[r] = 0.f; }

    unsigned short* pl = &plds[wave][0];
    const size_t gbase = ((size_t)(b * Hc + h)) * NQc * NKc
                       + (size_t)(qbase + 8 * hl) * NKc + ln;

    for (int kbase = 0; kbase < NKc; kbase += 32) {
        // ---- S = Q K^T for 2 key-halves of 16 ----
        v8f s[2];
        #pragma unroll
        for (int j = 0; j < 2; ++j) {
            const int key = kbase + j * 16 + ln;
            const unsigned short* krow =
                kb + ((size_t)(b * NKc + key)) * DM + h * Dc + 16 * hl;
            v8f z = {};
            z = wmma_bf16(aq[0], load16(krow), z);
            s[j] = wmma_bf16(aq[1], load16(krow + 32), z);
        }

        // ---- unconditional, clause-friendly geometry + mask loads (8 rows x 2 cols) ----
        float gv0[8], gv1[8], mneg0[8], mneg1[8];
        #pragma unroll
        for (int r = 0; r < 8; ++r) {
            const size_t gi = gbase + (size_t)r * NKc + kbase;
            __builtin_prefetch(geom + gi + 32, 0, 0);   // next key tile of this row
            gv0[r] = geom[gi];
            gv1[r] = geom[gi + 16];
            mneg0[r] = mask[gi]      ? NEG_INF : 0.0f;
            mneg1[r] = mask[gi + 16] ? NEG_INF : 0.0f;
        }

        // ---- branchless bias + online softmax ----
        #pragma unroll
        for (int r = 0; r < 8; ++r) {
            // additive -inf mask keeps log/loads unconditional (no exec branches)
            float w0 = s[0][r] * 0.125f + __logf(fmaxf(gv0[r], 1e-6f)) + mneg0[r];
            float w1 = s[1][r] * 0.125f + __logf(fmaxf(gv1[r], 1e-6f)) + mneg1[r];

            float tm = fmaxf(w0, w1);
            tm = fmaxf(tm, __shfl_xor(tm, 1));
            tm = fmaxf(tm, __shfl_xor(tm, 2));
            tm = fmaxf(tm, __shfl_xor(tm, 4));
            tm = fmaxf(tm, __shfl_xor(tm, 8));   // stays within 16-lane half

            float nm = fmaxf(mrow[r], tm);
            float sc = (mrow[r] == nm) ? 1.0f : __expf(mrow[r] - nm);
            float p0 = (w0 == NEG_INF) ? 0.f : __expf(w0 - nm);
            float p1 = (w1 == NEG_INF) ? 0.f : __expf(w1 - nm);
            float ps = p0 + p1;
            ps += __shfl_xor(ps, 1);
            ps += __shfl_xor(ps, 2);
            ps += __shfl_xor(ps, 4);
            ps += __shfl_xor(ps, 8);
            lrow[r] = lrow[r] * sc + ps;
            mrow[r] = nm;
            acc0[r] *= sc; acc1[r] *= sc; acc2[r] *= sc; acc3[r] *= sc;

            const int m = r + 8 * hl;
            pl[m * 32 + ln]      = f2bf(p0);     // stage P row-major for A-frag repack
            pl[m * 32 + 16 + ln] = f2bf(p1);
        }

        // ---- repack P (16x32) into WMMA A layout via LDS ----
        v16bf ap = load_afrag(pl + ln * 32, hl);

        // ---- O += P * V : 4 dv tiles of 16, per-lane contiguous keys from Vt ----
        const unsigned short* vcol =
            vt + ((size_t)(b * DM + h * Dc + ln)) * NKc + kbase + 16 * hl;
        acc0 = wmma_bf16(ap, load16(vcol), acc0);
        acc1 = wmma_bf16(ap, load16(vcol + (size_t)16 * NKc), acc1);
        acc2 = wmma_bf16(ap, load16(vcol + (size_t)32 * NKc), acc2);
        acc3 = wmma_bf16(ap, load16(vcol + (size_t)48 * NKc), acc3);
    }

    // ---- normalize + write O in (b, q, h*64+dv) layout ----
    #pragma unroll
    for (int r = 0; r < 8; ++r) {
        const int qrow = qbase + r + 8 * hl;
        float inv = (lrow[r] > 0.f) ? 1.0f / lrow[r] : 0.f;
        unsigned short* orow = ob + ((size_t)(b * NQc + qrow)) * DM + h * Dc + ln;
        orow[0]  = f2bf(acc0[r] * inv);
        orow[16] = f2bf(acc1[r] * inv);
        orow[32] = f2bf(acc2[r] * inv);
        orow[48] = f2bf(acc3[r] * inv);
    }
}

// ---------------- residual + LayerNorm (one wave per row) ----------------

__global__ __launch_bounds__(128) void ln_kernel(const float* __restrict__ x0,
                                                 const float* __restrict__ proj,
                                                 const float* __restrict__ gamma,
                                                 const float* __restrict__ beta,
                                                 float* __restrict__ out) {
    const int lane = threadIdx.x & 31;
    const int wave = threadIdx.x >> 5;
    const int row = blockIdx.x * 4 + wave;
    const float* xr = x0 + (size_t)row * DM;
    const float* pr = proj + (size_t)row * DM;

    float v[32], s = 0.f;
    #pragma unroll
    for (int i = 0; i < 32; ++i) { v[i] = xr[lane + 32 * i] + pr[lane + 32 * i]; s += v[i]; }
    #pragma unroll
    for (int m = 1; m < 32; m <<= 1) s += __shfl_xor(s, m);
    float mu = s * (1.0f / DM);
    float vs = 0.f;
    #pragma unroll
    for (int i = 0; i < 32; ++i) { float d = v[i] - mu; vs += d * d; }
    #pragma unroll
    for (int m = 1; m < 32; m <<= 1) vs += __shfl_xor(vs, m);
    float rstd = rsqrtf(vs * (1.0f / DM) + LN_EPS);
    #pragma unroll
    for (int i = 0; i < 32; ++i) {
        int c = lane + 32 * i;
        out[(size_t)row * DM + c] = (v[i] - mu) * rstd * gamma[c] + beta[c];
    }
}

// ---------------- host launch ----------------

extern "C" void kernel_launch(void* const* d_in, const int* in_sizes, int n_in,
                              void* d_out, int out_size, void* d_ws, size_t ws_size,
                              hipStream_t stream) {
    const float* queries = (const float*)d_in[0];
    const float* keys    = (const float*)d_in[1];
    const float* values  = (const float*)d_in[2];
    const float* geom    = (const float*)d_in[3];
    const unsigned char* mask = (const unsigned char*)d_in[4];
    const float* Wq = (const float*)d_in[5];
    const float* bq = (const float*)d_in[6];
    const float* Wk = (const float*)d_in[7];
    const float* bk = (const float*)d_in[8];
    const float* Wv = (const float*)d_in[9];
    const float* bv = (const float*)d_in[10];
    const float* Wo = (const float*)d_in[11];
    const float* bo = (const float*)d_in[12];
    const float* ln_g = (const float*)d_in[13];
    const float* ln_b = (const float*)d_in[14];

    const int M = Bc * NQc;          // 4096
    const int nAct = M * DM;         // 4M elems
    const int nW = DM * DM;          // 1M elems

    char* ws = (char*)d_ws;
    auto carve = [&](size_t bytes) -> char* {
        char* p = ws;
        ws += (bytes + 255) & ~(size_t)255;
        return p;
    };
    unsigned short* xq  = (unsigned short*)carve((size_t)nAct * 2);
    unsigned short* xk  = (unsigned short*)carve((size_t)nAct * 2);
    unsigned short* xv  = (unsigned short*)carve((size_t)nAct * 2);
    unsigned short* wqt = (unsigned short*)carve((size_t)nW * 2);
    unsigned short* wkt = (unsigned short*)carve((size_t)nW * 2);
    unsigned short* wvt = (unsigned short*)carve((size_t)nW * 2);
    unsigned short* wot = (unsigned short*)carve((size_t)nW * 2);
    unsigned short* qbf = (unsigned short*)carve((size_t)nAct * 2);
    unsigned short* kbf = (unsigned short*)carve((size_t)nAct * 2);
    unsigned short* vbf = (unsigned short*)carve((size_t)nAct * 2);
    unsigned short* vtb = (unsigned short*)carve((size_t)nAct * 2);
    unsigned short* obf = (unsigned short*)carve((size_t)nAct * 2);
    float* proj         = (float*)carve((size_t)nAct * 4);

    // 1) convert activations to bf16
    cvt_bf16_kernel<<<nAct / 4 / 256, 256, 0, stream>>>(queries, xq, nAct);
    cvt_bf16_kernel<<<nAct / 4 / 256, 256, 0, stream>>>(keys, xk, nAct);
    cvt_bf16_kernel<<<nAct / 4 / 256, 256, 0, stream>>>(values, xv, nAct);

    // 2) transpose-convert weights to bf16 [N,K]
    dim3 tb(32, 8), tg(DM / 32, DM / 32);
    transposeW_kernel<<<tg, tb, 0, stream>>>(Wq, wqt, DM, DM);
    transposeW_kernel<<<tg, tb, 0, stream>>>(Wk, wkt, DM, DM);
    transposeW_kernel<<<tg, tb, 0, stream>>>(Wv, wvt, DM, DM);
    transposeW_kernel<<<tg, tb, 0, stream>>>(Wo, wot, DM, DM);

    // 3) Q/K/V projections (WMMA, bf16 out)
    dim3 gg(DM / 64, M / 64);
    gemm_bf16_kernel<1><<<gg, 128, 0, stream>>>(xq, wqt, bq, qbf, M, DM, DM);
    gemm_bf16_kernel<1><<<gg, 128, 0, stream>>>(xk, wkt, bk, kbf, M, DM, DM);
    gemm_bf16_kernel<1><<<gg, 128, 0, stream>>>(xv, wvt, bv, vbf, M, DM, DM);

    // 4) V transpose for per-lane-contiguous PV B-fragments
    transposeV_kernel<<<dim3(DM / 32, NKc / 32, Bc), tb, 0, stream>>>(vbf, vtb);

    // 5) fused flash attention with geometry log-bias + mask
    attn_kernel<<<dim3(NQc / 64, Hc, Bc), 128, 0, stream>>>(qbf, kbf, vtb, geom, mask, obf);

    // 6) output projection (WMMA, fp32 out)
    gemm_bf16_kernel<0><<<gg, 128, 0, stream>>>(obf, wot, bo, proj, M, DM, DM);

    // 7) residual + LayerNorm
    ln_kernel<<<M / 4, 128, 0, stream>>>(queries, proj, ln_g, ln_b, (float*)d_out);
}
